// ChameleonVQVAEEncoderAttnBlock_32246614458385
// MI455X (gfx1250) — matrix operand order repaired and checked by reading, not compiled, not run on hardware
//
#include <hip/hip_runtime.h>
#include <hip/hip_bf16.h>

typedef __attribute__((ext_vector_type(16))) _Float16 v16h;
typedef __attribute__((ext_vector_type(8)))  float    v8f;
typedef __attribute__((ext_vector_type(8)))  _Float16 h8;
typedef __attribute__((ext_vector_type(4)))  _Float16 h4;
typedef __attribute__((ext_vector_type(4)))  float    f4;

// Problem constants: B=16, C=512, H=W=32, N=1024, 32 groups (16 ch/group).
#define BATCH 16
#define CDIM  512
#define NPIX  1024
#define NGRP  32
#define GSZ   16

#define OUT_T16 0   // f16, transposed [n][o] (vectorized h8 stores)
#define OUT_N16 1   // f16, standard   [o][n]
#define OUT_F32 2   // f32, standard + residual

#define WMMA_F16(A, Bm, C) \
  __builtin_amdgcn_wmma_f32_16x16x32_f16(false, (A), false, (Bm), (short)0, (C), false, false)

// ---------------------------------------------------------------------------
// WMMA fragment helpers (CDNA5 ISA 7.12.2, wave32).
__device__ __forceinline__ v16h load_a_frag(const _Float16* lds, int row_base,
                                            int lane) {
  const int m  = row_base + (lane & 15);
  const int kb = ((lane >> 4) & 1) * 8;
  const _Float16* rp = lds + m * 32;
  v16h a;
#pragma unroll
  for (int i = 0; i < 8; ++i) { a[i] = rp[kb + i]; a[8 + i] = rp[kb + 16 + i]; }
  return a;
}
__device__ __forceinline__ v16h load_bT_frag(const _Float16* ldsT, int col_base,
                                             int lane) {
  const int n  = col_base + (lane & 15);
  const int kb = ((lane >> 4) & 1) * 8;
  const _Float16* rp = ldsT + n * 32;
  v16h b;
#pragma unroll
  for (int i = 0; i < 8; ++i) { b[i] = rp[kb + i]; b[8 + i] = rp[kb + 16 + i]; }
  return b;
}

// ---------------------------------------------------------------------------
// Kernel 1: GroupNorm -> TRANSPOSED f16 activations hT[b][n][c].
// In-LDS transpose so the global write is 2x b128 of 16 contiguous channels.
__global__ __launch_bounds__(256) void groupnorm_t_kernel(
    const float* __restrict__ x, const float* __restrict__ gamma,
    const float* __restrict__ beta, _Float16* __restrict__ hT) {
  const int bg = blockIdx.x;  // b*NGRP + g
  const int b = bg >> 5, g = bg & 31;
  const size_t base = ((size_t)b * CDIM + g * GSZ) * NPIX;
  const int t = threadIdx.x;
  const int TOTV = GSZ * NPIX / 4;  // 4096 float4s

  const f4* xv = (const f4*)(x + base);
  float s = 0.f, ss = 0.f;
  for (int i = t; i < TOTV; i += 256) {
    f4 v = xv[i];
#pragma unroll
    for (int e = 0; e < 4; ++e) { s += v[e]; ss += v[e] * v[e]; }
  }
  __shared__ float r1[256], r2[256];
  r1[t] = s; r2[t] = ss; __syncthreads();
  for (int k = 128; k; k >>= 1) {
    if (t < k) { r1[t] += r1[t + k]; r2[t] += r2[t + k]; }
    __syncthreads();
  }
  const float mean = r1[0] * (1.0f / (GSZ * NPIX));
  const float var  = r2[0] * (1.0f / (GSZ * NPIX)) - mean * mean;
  const float inv  = rsqrtf(var + 1e-6f);

  // Transpose pass: thread owns local channel c = t>>4, n-span ns..ns+15.
  const int c  = t >> 4;            // 0..15
  const int ns = (t & 15) * 16;     // 0..240 within a 256-pixel chunk
  const float gm = gamma[g * GSZ + c] * inv;
  const float bt = beta[g * GSZ + c] - mean * gm;

  __shared__ __align__(16) _Float16 Ts[256 * GSZ];  // [n_local][c] (8 KB)
  _Float16* outp = hT + (size_t)b * NPIX * CDIM + (size_t)g * GSZ;

  for (int ch = 0; ch < NPIX; ch += 256) {
    __syncthreads();
    const float* xp = x + base + (size_t)c * NPIX + ch + ns;
    f4 v0 = *(const f4*)(xp),     v1 = *(const f4*)(xp + 4),
       v2 = *(const f4*)(xp + 8), v3 = *(const f4*)(xp + 12);
#pragma unroll
    for (int e = 0; e < 4; ++e) {
      Ts[(ns + e)      * GSZ + c] = (_Float16)(v0[e] * gm + bt);
      Ts[(ns + 4 + e)  * GSZ + c] = (_Float16)(v1[e] * gm + bt);
      Ts[(ns + 8 + e)  * GSZ + c] = (_Float16)(v2[e] * gm + bt);
      Ts[(ns + 12 + e) * GSZ + c] = (_Float16)(v3[e] * gm + bt);
    }
    __syncthreads();
    // write-out: thread t -> pixel n = ch + t, 16 contiguous channels
    h8 o0 = *(const h8*)&Ts[t * GSZ];
    h8 o1 = *(const h8*)&Ts[t * GSZ + 8];
    _Float16* op = outp + (size_t)(ch + t) * CDIM;
    *(h8*)op       = o0;
    *(h8*)(op + 8) = o1;
  }
}

// ---------------------------------------------------------------------------
// 1x1 conv GEMM  D[o,n] = sum_c W[o,c]*HT[n,c] + bias[o]
// Activations always [b][n][c] -> both staging paths are contiguous b128.
// 64x64 block, 2x2 wave grid, software-pipelined staging.
template <int OMODE>
__global__ __launch_bounds__(128) void gemm_conv_kernel(
    const float* __restrict__ W, const float* __restrict__ bias,
    const _Float16* __restrict__ HT,    // [B][N][C]
    _Float16* __restrict__ Out16, float* __restrict__ Out32,
    const float* __restrict__ Residual) {
  const int o0 = blockIdx.x * 64, n0 = blockIdx.y * 64, b = blockIdx.z;
  const int tid = threadIdx.x, lane = tid & 31, wv = tid >> 5;
  const int wr = (wv >> 1) * 32, wc = (wv & 1) * 32;
  __shared__ __align__(16) _Float16 As[64 * 32];   // [m=o][k=c]
  __shared__ __align__(16) _Float16 BsT[64 * 32];  // [n][k=c]
  v8f acc[2][2] = {};
  const int mm = tid >> 1, ks = (tid & 1) * 16;    // 2 threads/row

  f4 w0, w1, w2, w3;
  h8 hb0, hb1;
  auto fetch = [&](int k0) {
    const float* wp = W + (size_t)(o0 + mm) * CDIM + (k0 + ks);
    w0 = *(const f4*)(wp);     w1 = *(const f4*)(wp + 4);
    w2 = *(const f4*)(wp + 8); w3 = *(const f4*)(wp + 12);
    const _Float16* hp = HT + ((size_t)b * NPIX + (n0 + mm)) * CDIM + (k0 + ks);
    hb0 = *(const h8*)hp; hb1 = *(const h8*)(hp + 8);
  };
  auto stage = [&]() {
    h8 a0, a1;
#pragma unroll
    for (int e = 0; e < 4; ++e) {
      a0[e] = (_Float16)w0[e]; a0[e + 4] = (_Float16)w1[e];
      a1[e] = (_Float16)w2[e]; a1[e + 4] = (_Float16)w3[e];
    }
    *(h8*)&As[mm * 32 + ks]      = a0;
    *(h8*)&As[mm * 32 + ks + 8]  = a1;
    *(h8*)&BsT[mm * 32 + ks]     = hb0;
    *(h8*)&BsT[mm * 32 + ks + 8] = hb1;
  };

  fetch(0);
  for (int k0 = 0; k0 < CDIM; k0 += 32) {
    stage();
    __syncthreads();
    if (k0 + 32 < CDIM) fetch(k0 + 32);  // overlap next global load with WMMA
    v16h af0 = load_a_frag(As, wr, lane);
    v16h af1 = load_a_frag(As, wr + 16, lane);
    v16h bf0 = load_bT_frag(BsT, wc, lane);
    v16h bf1 = load_bT_frag(BsT, wc + 16, lane);
    acc[0][0] = WMMA_F16(af0, bf0, acc[0][0]);
    acc[0][1] = WMMA_F16(af0, bf1, acc[0][1]);
    acc[1][0] = WMMA_F16(af1, bf0, acc[1][0]);
    acc[1][1] = WMMA_F16(af1, bf1, acc[1][1]);
    __syncthreads();
  }
  const int n_lo = lane & 15, mhi = ((lane >> 4) & 1) * 8;
  if (OMODE == OUT_T16) {
    // lane holds 8 consecutive o for fixed n -> contiguous in [n][o] layout
#pragma unroll
    for (int mi = 0; mi < 2; ++mi)
#pragma unroll
      for (int ni = 0; ni < 2; ++ni) {
        int ob = o0 + wr + mi * 16 + mhi;
        int n  = n0 + wc + ni * 16 + n_lo;
        h8 ov;
#pragma unroll
        for (int r = 0; r < 8; ++r)
          ov[r] = (_Float16)(acc[mi][ni][r] + bias[ob + r]);
        *(h8*)&Out16[((size_t)b * NPIX + n) * CDIM + ob] = ov;
      }
  } else {
#pragma unroll
    for (int mi = 0; mi < 2; ++mi)
#pragma unroll
      for (int ni = 0; ni < 2; ++ni)
#pragma unroll
        for (int r = 0; r < 8; ++r) {
          int o = o0 + wr + mi * 16 + mhi + r;
          int n = n0 + wc + ni * 16 + n_lo;
          float v = acc[mi][ni][r] + bias[o];
          size_t oi = ((size_t)b * CDIM + o) * NPIX + n;
          if (OMODE == OUT_F32) Out32[oi] = Residual[oi] + v;
          else                  Out16[oi] = (_Float16)v;
        }
  }
}

// ---------------------------------------------------------------------------
// Kernel 3: S[b,i,j] = scale * sum_c QT[b,i,c]*KT[b,j,c]
__global__ __launch_bounds__(128) void attn_scores_kernel(
    const _Float16* __restrict__ QT, const _Float16* __restrict__ KT,
    float* __restrict__ S) {
  const float scale = 0.04419417382415922f;  // 512^-0.5
  const int i0 = blockIdx.x * 64, j0 = blockIdx.y * 64, b = blockIdx.z;
  const int tid = threadIdx.x, lane = tid & 31, wv = tid >> 5;
  const int wr = (wv >> 1) * 32, wc = (wv & 1) * 32;
  __shared__ __align__(16) _Float16 As[64 * 32];   // [i][c]
  __shared__ __align__(16) _Float16 BsT[64 * 32];  // [j][c]
  v8f acc[2][2] = {};
  const int mm = tid >> 1, ks = (tid & 1) * 16;

  h8 q0, q1, c0, c1;
  auto fetch = [&](int k0) {
    const _Float16* qp = QT + ((size_t)b * NPIX + (i0 + mm)) * CDIM + (k0 + ks);
    q0 = *(const h8*)qp; q1 = *(const h8*)(qp + 8);
    const _Float16* kp = KT + ((size_t)b * NPIX + (j0 + mm)) * CDIM + (k0 + ks);
    c0 = *(const h8*)kp; c1 = *(const h8*)(kp + 8);
  };
  auto stage = [&]() {
    *(h8*)&As[mm * 32 + ks]      = q0;
    *(h8*)&As[mm * 32 + ks + 8]  = q1;
    *(h8*)&BsT[mm * 32 + ks]     = c0;
    *(h8*)&BsT[mm * 32 + ks + 8] = c1;
  };

  fetch(0);
  for (int k0 = 0; k0 < CDIM; k0 += 32) {
    stage();
    __syncthreads();
    if (k0 + 32 < CDIM) fetch(k0 + 32);
    v16h af0 = load_a_frag(As, wr, lane);
    v16h af1 = load_a_frag(As, wr + 16, lane);
    v16h bf0 = load_bT_frag(BsT, wc, lane);
    v16h bf1 = load_bT_frag(BsT, wc + 16, lane);
    acc[0][0] = WMMA_F16(af0, bf0, acc[0][0]);
    acc[0][1] = WMMA_F16(af0, bf1, acc[0][1]);
    acc[1][0] = WMMA_F16(af1, bf0, acc[1][0]);
    acc[1][1] = WMMA_F16(af1, bf1, acc[1][1]);
    __syncthreads();
  }
  const int n_lo = lane & 15, mhi = ((lane >> 4) & 1) * 8;
#pragma unroll
  for (int mi = 0; mi < 2; ++mi)
#pragma unroll
    for (int ni = 0; ni < 2; ++ni)
#pragma unroll
      for (int r = 0; r < 8; ++r) {
        int i = i0 + wr + mi * 16 + mhi + r;
        int j = j0 + wc + ni * 16 + n_lo;
        S[((size_t)b * NPIX + i) * NPIX + j] = acc[mi][ni][r] * scale;
      }
}

// ---------------------------------------------------------------------------
// Kernel 4: row softmax; writes packed f16 P over the FRONT of the same row.
__global__ __launch_bounds__(256) void softmax_kernel(float* __restrict__ S) {
  const int row = blockIdx.x;  // b*N + i
  float* rp = S + (size_t)row * NPIX;
  const int t = threadIdx.x;
  __shared__ float red[256];
  f4 v = *((const f4*)rp + t);
  float m = fmaxf(fmaxf(v[0], v[1]), fmaxf(v[2], v[3]));
  red[t] = m; __syncthreads();
  for (int k = 128; k; k >>= 1) {
    if (t < k) red[t] = fmaxf(red[t], red[t + k]);
    __syncthreads();
  }
  m = red[0]; __syncthreads();
  float s = 0.f;
#pragma unroll
  for (int e = 0; e < 4; ++e) { v[e] = __expf(v[e] - m); s += v[e]; }
  red[t] = s; __syncthreads();
  for (int k = 128; k; k >>= 1) {
    if (t < k) red[t] += red[t + k];
    __syncthreads();
  }
  const float inv = 1.0f / red[0];
  __syncthreads();
  h4 o;
#pragma unroll
  for (int e = 0; e < 4; ++e) o[e] = (_Float16)(v[e] * inv);
  *((h4*)rp + t) = o;
}

// ---------------------------------------------------------------------------
// Kernel 5: OT[b,i,c] = sum_j V[b,c,j]*P[b,i,j]; transposed h8 epilogue.
__global__ __launch_bounds__(128) void attn_av_kernel(
    const _Float16* __restrict__ V, const float* __restrict__ Sbase,
    _Float16* __restrict__ OT) {
  const int c0 = blockIdx.x * 64, i0 = blockIdx.y * 64, b = blockIdx.z;
  const int tid = threadIdx.x, lane = tid & 31, wv = tid >> 5;
  const int wr = (wv >> 1) * 32, wc = (wv & 1) * 32;
  __shared__ __align__(16) _Float16 As[64 * 32];   // [c][j]
  __shared__ __align__(16) _Float16 BsT[64 * 32];  // [i][j]
  v8f acc[2][2] = {};
  const _Float16* Vb = V + (size_t)b * CDIM * NPIX;
  const int mm = tid >> 1, ks = (tid & 1) * 16;

  h8 a0r, a1r, b0r, b1r;
  auto fetch = [&](int k0) {
    const _Float16* vp = Vb + (size_t)(c0 + mm) * NPIX + (k0 + ks);
    a0r = *(const h8*)vp; a1r = *(const h8*)(vp + 8);
    const _Float16* pp =
        (const _Float16*)(Sbase + (size_t)(b * NPIX + i0 + mm) * NPIX) +
        (k0 + ks);
    b0r = *(const h8*)pp; b1r = *(const h8*)(pp + 8);
  };
  auto stage = [&]() {
    *(h8*)&As[mm * 32 + ks]      = a0r;
    *(h8*)&As[mm * 32 + ks + 8]  = a1r;
    *(h8*)&BsT[mm * 32 + ks]     = b0r;
    *(h8*)&BsT[mm * 32 + ks + 8] = b1r;
  };

  fetch(0);
  for (int k0 = 0; k0 < NPIX; k0 += 32) {
    stage();
    __syncthreads();
    if (k0 + 32 < NPIX) fetch(k0 + 32);
    v16h af0 = load_a_frag(As, wr, lane);
    v16h af1 = load_a_frag(As, wr + 16, lane);
    v16h bf0 = load_bT_frag(BsT, wc, lane);
    v16h bf1 = load_bT_frag(BsT, wc + 16, lane);
    acc[0][0] = WMMA_F16(af0, bf0, acc[0][0]);
    acc[0][1] = WMMA_F16(af0, bf1, acc[0][1]);
    acc[1][0] = WMMA_F16(af1, bf0, acc[1][0]);
    acc[1][1] = WMMA_F16(af1, bf1, acc[1][1]);
    __syncthreads();
  }
  const int n_lo = lane & 15, mhi = ((lane >> 4) & 1) * 8;
#pragma unroll
  for (int mi = 0; mi < 2; ++mi)
#pragma unroll
    for (int ni = 0; ni < 2; ++ni) {
      int cb = c0 + wr + mi * 16 + mhi;  // 8 consecutive c
      int i  = i0 + wc + ni * 16 + n_lo;
      h8 ov;
#pragma unroll
      for (int r = 0; r < 8; ++r) ov[r] = (_Float16)acc[mi][ni][r];
      *(h8*)&OT[((size_t)b * NPIX + i) * CDIM + cb] = ov;
    }
}

// ---------------------------------------------------------------------------
extern "C" void kernel_launch(void* const* d_in, const int* in_sizes, int n_in,
                              void* d_out, int out_size, void* d_ws,
                              size_t ws_size, hipStream_t stream) {
  const float* x     = (const float*)d_in[0];
  const float* gamma = (const float*)d_in[1];
  const float* beta  = (const float*)d_in[2];
  const float* wq = (const float*)d_in[3]; const float* bq = (const float*)d_in[4];
  const float* wk = (const float*)d_in[5]; const float* bk = (const float*)d_in[6];
  const float* wv = (const float*)d_in[7]; const float* bv = (const float*)d_in[8];
  const float* wo = (const float*)d_in[9]; const float* bo = (const float*)d_in[10];
  float* out = (float*)d_out;

  // Workspace: qT|kT|v16|hT(->oT)  (4 x 16 MB)  then S (64 MB, fits L2)
  char* ws = (char*)d_ws;
  const size_t HB = (size_t)BATCH * CDIM * NPIX * sizeof(_Float16);
  _Float16* qT  = (_Float16*)(ws);            // [B][N][C]
  _Float16* kT  = (_Float16*)(ws + HB);       // [B][N][C]
  _Float16* v16 = (_Float16*)(ws + 2 * HB);   // [B][C][N]
  _Float16* hT  = (_Float16*)(ws + 3 * HB);   // [B][N][C]; reused as oT
  float*    S   = (float*)(ws + 4 * HB);

  groupnorm_t_kernel<<<BATCH * NGRP, 256, 0, stream>>>(x, gamma, beta, hT);

  dim3 gc(CDIM / 64, NPIX / 64, BATCH);          // 8 x 16 x 16
  gemm_conv_kernel<OUT_T16><<<gc, 128, 0, stream>>>(wq, bq, hT, qT, nullptr, nullptr);
  gemm_conv_kernel<OUT_T16><<<gc, 128, 0, stream>>>(wk, bk, hT, kT, nullptr, nullptr);
  gemm_conv_kernel<OUT_N16><<<gc, 128, 0, stream>>>(wv, bv, hT, v16, nullptr, nullptr);

  dim3 gs(NPIX / 64, NPIX / 64, BATCH);          // 16 x 16 x 16
  attn_scores_kernel<<<gs, 128, 0, stream>>>(qT, kT, S);
  softmax_kernel<<<BATCH * NPIX, 256, 0, stream>>>(S);
  attn_av_kernel<<<gc, 128, 0, stream>>>(v16, S, hT /* oT */);

  gemm_conv_kernel<OUT_F32><<<gc, 128, 0, stream>>>(wo, bo, hT, nullptr, out, x);
}
